// LeWinTransformerBlock_73693048865421
// MI455X (gfx1250) — matrix operand order repaired and checked by reading, not compiled, not run on hardware
//
#include <hip/hip_runtime.h>
#include <hip/hip_bf16.h>
#include <math.h>

// ---------------------------------------------------------------------------
// LeWin transformer block for MI455X (gfx1250, wave32, WMMA 16x16x32 f16)
// ---------------------------------------------------------------------------
constexpr int Bn   = 2;
constexpr int Hh   = 512;
constexpr int Ww   = 512;
constexpr int Cc   = 32;
constexpr int WIN  = 8;
constexpr int Nt   = WIN * WIN;          // 64 tokens per window
constexpr int HID  = 128;
constexpr float SCALEQ = 0.17677669529663687f;  // 32^-0.5
constexpr float EPS    = 1e-3f;
constexpr float SLOPE  = 0.3f;

typedef __attribute__((ext_vector_type(16))) _Float16 v16h;
typedef __attribute__((ext_vector_type(8)))  _Float16 v8h;   // 16 bytes
typedef __attribute__((ext_vector_type(8)))  float    v8f;

#if __has_builtin(__builtin_amdgcn_global_load_async_to_lds_b128)
#define ASYNC_COPY 1
// builtin signature (leaked by round-2 diagnostic): first param is
// 'int __vector(4) __device__ *'  ==  addrspace(1) pointer to vector_size(16) int
typedef int vint4 __attribute__((vector_size(4 * sizeof(int))));
typedef __attribute__((address_space(1))) vint4 gvint4;
typedef __attribute__((address_space(3))) vint4 lvint4;
#else
#define ASYNC_COPY 0
#endif

// ---- WMMA fragment loaders (ISA 7.12.2 layouts, wave32) --------------------
// A matrix 16x32 f16 row-major [m][k] at base[m*ld+k]:
//   lane(m = lane&15, h = lane>>4) holds k = 16g + 8h + (0..7)  for g = 0,1
//   -> two contiguous 16B runs per lane (two ds_load_b128).
__device__ __forceinline__ v16h ldA(const _Float16* base, int ld, int lane) {
  const int m = lane & 15, h = lane >> 4;
  const v8h lo = *(const v8h*)(base + m * ld + h * 8);
  const v8h hi = *(const v8h*)(base + m * ld + 16 + h * 8);
  v16h a;
#pragma unroll
  for (int j = 0; j < 8; ++j) { a[j] = lo[j]; a[j + 8] = hi[j]; }
  return a;
}
// B matrix 32x16 f16 where storage is TRANSPOSED: element B[k][n] at
// base[n*ld + k]. Lane(n = lane&15, h = lane>>4) holds k = 16g + 8h + (0..7) for g = 0,1
//   -> two contiguous 16B runs per lane (two ds_load_b128).
__device__ __forceinline__ v16h ldBT(const _Float16* base, int ld, int lane) {
  const int n = lane & 15, h = lane >> 4;
  const v8h lo = *(const v8h*)(base + n * ld + h * 8);
  const v8h hi = *(const v8h*)(base + n * ld + 16 + h * 8);
  v16h b;
#pragma unroll
  for (int j = 0; j < 8; ++j) { b[j] = lo[j]; b[j + 8] = hi[j]; }
  return b;
}

__device__ __forceinline__ v8f wmma16(v16h a, v16h b, v8f c) {
  return __builtin_amdgcn_wmma_f32_16x16x32_f16(false, a, false, b,
                                                (short)0, c, false, false);
}

__device__ __forceinline__ float gelu_exact(float z) {
  return 0.5f * z * (1.0f + erff(z * 0.70710678118654752f));
}

// ---------------------------------------------------------------------------
// Kernel 1: LN1 + window attention + proj + residual  ->  x2 (in d_out)
// one block (128 thr = 4 wave32) per 8x8 window
// ---------------------------------------------------------------------------
__global__ __launch_bounds__(128) void kwin_attn(
    const float* __restrict__ x,  const float* __restrict__ g1,
    const float* __restrict__ be1,
    const float* __restrict__ Wq, const float* __restrict__ bq,
    const float* __restrict__ Wkv,const float* __restrict__ bkv,
    const float* __restrict__ bias_table, const float* __restrict__ Wp,
    const float* __restrict__ bp, const int* __restrict__ rel_idx,
    float* __restrict__ x2) {
  __shared__ __attribute__((aligned(16))) _Float16 xnH[Nt * Cc];
  __shared__ __attribute__((aligned(16))) _Float16 qH [Nt * Cc];
  __shared__ __attribute__((aligned(16))) _Float16 kH [Nt * Cc];   // [token][d]
  __shared__ __attribute__((aligned(16))) _Float16 vT [Cc * Nt];   // [d][token]
  __shared__ __attribute__((aligned(16))) float    attnF[Nt * 65]; // padded rows
  __shared__ __attribute__((aligned(16))) _Float16 attnH[Nt * Nt];
  __shared__ __attribute__((aligned(16))) _Float16 outH[Nt * Cc];
  __shared__ __attribute__((aligned(16))) _Float16 WqT [Cc * Cc];  // [out][in]
  __shared__ __attribute__((aligned(16))) _Float16 WkvT[2 * Cc * Cc];
  __shared__ __attribute__((aligned(16))) _Float16 WpT [Cc * Cc];
  __shared__ float bqL[Cc], bkvL[2 * Cc], bpL[Cc];
  __shared__ __attribute__((aligned(16))) float shortF[Nt * Cc];

  const int tid  = threadIdx.x;
  const int lane = tid & 31;
  const int wave = tid >> 5;
  const int row0 = wave << 4;

  const int wi   = blockIdx.x;           // 0..8191
  const int b    = wi >> 12;
  const int widx = wi & 4095;
  const int wy   = widx >> 6;
  const int wx   = widx & 63;

  // token -> global mapping for this thread (also used for LN)
  long long tokBase = 0;
  if (tid < Nt) {
    const int gy = wy * WIN + (tid >> 3);
    const int gx = wx * WIN + (tid & 7);
    tokBase = ((long long)b * (Hh * Ww) + (long long)gy * Ww + gx) * Cc;
    __builtin_prefetch(x + tokBase, 0, 1);     // global_prefetch_b8
  }

  // weights -> LDS, transposed so B fragments are contiguous per lane
  for (int i = tid; i < Cc * Cc; i += 128)
    WqT[i] = (_Float16)Wq[(i & 31) * Cc + (i >> 5)];
  for (int i = tid; i < 2 * Cc * Cc; i += 128)
    WkvT[i] = (_Float16)Wkv[(i & 31) * (2 * Cc) + (i >> 5)];
  for (int i = tid; i < Cc * Cc; i += 128)
    WpT[i] = (_Float16)Wp[(i & 31) * Cc + (i >> 5)];
  if (tid < Cc)     bqL[tid]  = bq[tid];
  if (tid < 2 * Cc) bkvL[tid] = bkv[tid];
  if (tid < Cc)     bpL[tid]  = bp[tid];

  // ---- LayerNorm1 per token -> xnH, keep shortcut ----
  if (tid < Nt) {
    float vbuf[Cc];
    const float4* xv = (const float4*)(x + tokBase);
    float m = 0.f;
#pragma unroll
    for (int c4 = 0; c4 < Cc / 4; ++c4) {
      const float4 f = xv[c4];
      vbuf[4 * c4 + 0] = f.x; vbuf[4 * c4 + 1] = f.y;
      vbuf[4 * c4 + 2] = f.z; vbuf[4 * c4 + 3] = f.w;
      m += f.x + f.y + f.z + f.w;
    }
    m *= (1.f / Cc);
    float var = 0.f;
#pragma unroll
    for (int c = 0; c < Cc; ++c) { float d = vbuf[c] - m; var += d * d; }
    var *= (1.f / Cc);
    const float inv = rsqrtf(var + EPS);
#pragma unroll
    for (int c4 = 0; c4 < Cc / 4; ++c4) {
      float4 s;
      s.x = vbuf[4*c4+0]; s.y = vbuf[4*c4+1]; s.z = vbuf[4*c4+2]; s.w = vbuf[4*c4+3];
      *(float4*)&shortF[tid * Cc + 4 * c4] = s;
    }
#pragma unroll
    for (int c = 0; c < Cc; ++c)
      xnH[tid * Cc + c] = (_Float16)(((vbuf[c] - m) * inv) * g1[c] + be1[c]);
  }
  __syncthreads();

  const int fn = lane & 15, fh = lane >> 4;

  // ---- Q / KV projections (leaky-activated) ----
  {
    const v16h a = ldA(&xnH[row0 * Cc], Cc, lane);
#pragma unroll
    for (int nt = 0; nt < 2; ++nt) {          // Q: 64x32, store q*scale
      const v16h bw = ldBT(&WqT[(nt * 16) * Cc], Cc, lane);
      v8f acc = {};
      acc = wmma16(a, bw, acc);
      const int cidx = nt * 16 + fn;
#pragma unroll
      for (int r = 0; r < 8; ++r) {
        float v = acc[r] + bqL[cidx];
        v = (v >= 0.f) ? v : SLOPE * v;
        qH[(row0 + r + 8 * fh) * Cc + cidx] = (_Float16)(v * SCALEQ);
      }
    }
#pragma unroll
    for (int nt = 0; nt < 4; ++nt) {          // KV: k = ch 0..31 -> kH, v -> vT
      const v16h bw = ldBT(&WkvT[(nt * 16) * Cc], Cc, lane);
      v8f acc = {};
      acc = wmma16(a, bw, acc);
      const int cidx = nt * 16 + fn;
#pragma unroll
      for (int r = 0; r < 8; ++r) {
        float v = acc[r] + bkvL[cidx];
        v = (v >= 0.f) ? v : SLOPE * v;
        const int m = row0 + r + 8 * fh;
        if (nt < 2) kH[m * Cc + cidx]             = (_Float16)v;
        else        vT[(cidx - Cc) * Nt + m]      = (_Float16)v;  // transposed
      }
    }
  }
  __syncthreads();

  // ---- attn = (Q*scale) @ K^T  (kH already "transposed B": [token][d]) ----
  {
    const v16h aq = ldA(&qH[row0 * Cc], Cc, lane);
#pragma unroll
    for (int nt = 0; nt < 4; ++nt) {
      const v16h bk = ldBT(&kH[(nt * 16) * Cc], Cc, lane);
      v8f acc = {};
      acc = wmma16(aq, bk, acc);
#pragma unroll
      for (int r = 0; r < 8; ++r)
        attnF[(row0 + r + 8 * fh) * 65 + nt * 16 + fn] = acc[r];
    }
  }
  __syncthreads();

  // ---- relative-position bias + softmax (one row per thread) ----
  if (tid < Nt) {
    const int t = tid;
    float mx = -1e30f;
    for (int j = 0; j < Nt; ++j) {
      const float s = attnF[t * 65 + j] + bias_table[rel_idx[t * Nt + j]];
      attnF[t * 65 + j] = s;
      mx = fmaxf(mx, s);
    }
    float sum = 0.f;
    for (int j = 0; j < Nt; ++j) {
      const float e = expf(attnF[t * 65 + j] - mx);
      attnF[t * 65 + j] = e;
      sum += e;
    }
    const float inv = 1.f / sum;
    for (int j = 0; j < Nt; ++j)
      attnH[t * Nt + j] = (_Float16)(attnF[t * 65 + j] * inv);
  }
  __syncthreads();

  // ---- out = attn @ V  (vT is [d][token], perfect transposed-B) ----
#pragma unroll
  for (int nt = 0; nt < 2; ++nt) {
    v8f acc = {};
#pragma unroll
    for (int kc = 0; kc < 2; ++kc) {
      const v16h aa = ldA(&attnH[row0 * Nt + kc * 32], Nt, lane);
      const v16h bb = ldBT(&vT[(nt * 16) * Nt + kc * 32], Nt, lane);
      acc = wmma16(aa, bb, acc);
    }
#pragma unroll
    for (int r = 0; r < 8; ++r)
      outH[(row0 + r + 8 * fh) * Cc + nt * 16 + fn] = (_Float16)acc[r];
  }
  __syncthreads();

  // ---- proj + bias + residual -> f32 staging (reuse attnF region) ----
  float* outF = attnF;   // attnF dead after softmax; 2048 floats needed
  {
    const v16h ao = ldA(&outH[row0 * Cc], Cc, lane);
#pragma unroll
    for (int nt = 0; nt < 2; ++nt) {
      const v16h bw = ldBT(&WpT[(nt * 16) * Cc], Cc, lane);
      v8f acc = {};
      acc = wmma16(ao, bw, acc);
      const int cidx = nt * 16 + fn;
#pragma unroll
      for (int r = 0; r < 8; ++r) {
        const int m = row0 + r + 8 * fh;
        outF[m * Cc + cidx] = acc[r] + bpL[cidx] + shortF[m * Cc + cidx];
      }
    }
  }
  __syncthreads();

  // ---- cooperative coalesced 16B stores of the x2 tile ----
  for (int i = tid; i < Nt * Cc / 4; i += 128) {   // 512 float4
    const int t  = i >> 3, c4 = i & 7;
    const int gy = wy * WIN + (t >> 3);
    const int gx = wx * WIN + (t & 7);
    const long long gaddr =
        ((long long)b * (Hh * Ww) + (long long)gy * Ww + gx) * Cc + c4 * 4;
    *(float4*)&x2[gaddr] = *(const float4*)&outF[t * Cc + c4 * 4];
  }
}

// ---------------------------------------------------------------------------
// Kernel 2: LN2 + W1 (32->128) + GELU  ->  h (f16, in d_ws)
// one block per 64 tokens
// ---------------------------------------------------------------------------
__global__ __launch_bounds__(128) void kmlp1(
    const float* __restrict__ x2, const float* __restrict__ g2,
    const float* __restrict__ be2, const float* __restrict__ W1,
    const float* __restrict__ b1, _Float16* __restrict__ hOut) {
  __shared__ __attribute__((aligned(16))) _Float16 ynH[Nt * Cc];
  __shared__ __attribute__((aligned(16))) _Float16 W1T[HID * Cc];  // [out][in]
  __shared__ __attribute__((aligned(16))) _Float16 hh [Nt * HID];
  __shared__ float b1L[HID];

  const int tid = threadIdx.x, lane = tid & 31, wave = tid >> 5;
  const int row0 = wave << 4;
  const long long t0 = (long long)blockIdx.x * Nt;

  if (tid < Nt) __builtin_prefetch(x2 + (t0 + tid) * Cc, 0, 1);

  for (int i = tid; i < HID * Cc; i += 128)
    W1T[i] = (_Float16)W1[(i & 31) * HID + (i >> 5)];
  b1L[tid] = b1[tid];  // blockDim 128 == HID

  if (tid < Nt) {
    const float4* xv = (const float4*)(x2 + (t0 + tid) * Cc);
    float vbuf[Cc];
    float m = 0.f;
#pragma unroll
    for (int c4 = 0; c4 < Cc / 4; ++c4) {
      const float4 f = xv[c4];
      vbuf[4 * c4 + 0] = f.x; vbuf[4 * c4 + 1] = f.y;
      vbuf[4 * c4 + 2] = f.z; vbuf[4 * c4 + 3] = f.w;
      m += f.x + f.y + f.z + f.w;
    }
    m *= (1.f / Cc);
    float var = 0.f;
#pragma unroll
    for (int c = 0; c < Cc; ++c) { float d = vbuf[c] - m; var += d * d; }
    var *= (1.f / Cc);
    const float inv = rsqrtf(var + EPS);
#pragma unroll
    for (int c = 0; c < Cc; ++c)
      ynH[tid * Cc + c] = (_Float16)(((vbuf[c] - m) * inv) * g2[c] + be2[c]);
  }
  __syncthreads();

  const int fn = lane & 15, fh = lane >> 4;
  const v16h a = ldA(&ynH[row0 * Cc], Cc, lane);
#pragma unroll
  for (int nt = 0; nt < 8; ++nt) {
    const v16h bw = ldBT(&W1T[(nt * 16) * Cc], Cc, lane);
    v8f acc = {};
    acc = wmma16(a, bw, acc);
    const int cidx = nt * 16 + fn;
#pragma unroll
    for (int r = 0; r < 8; ++r) {
      const float z = acc[r] + b1L[cidx];
      hh[(row0 + r + 8 * fh) * HID + cidx] = (_Float16)gelu_exact(z);
    }
  }
  __syncthreads();

  // cooperative coalesced 16B stores (this is the 134MB buffer: keep it fast)
  for (int i = tid; i < Nt * HID / 8; i += 128) {  // 1024 x 16B
    const int tok = i >> 4, part = i & 15;
    *(uint4*)(hOut + (t0 + tok) * HID + part * 8) = *(const uint4*)&hh[i * 8];
  }
}

// ---------------------------------------------------------------------------
// Kernel 3: 3x3 depthwise conv + GELU + W2 (128->32) + residual -> d_out
// one block per 8x8 spatial tile; 10x10x128 f16 halo in LDS (async if avail)
// ---------------------------------------------------------------------------
__global__ __launch_bounds__(128) void kmlp2(
    const _Float16* __restrict__ hIn, const float* __restrict__ dwk,
    const float* __restrict__ dwb, const float* __restrict__ W2,
    const float* __restrict__ b2, float* __restrict__ out) {
  __shared__ __attribute__((aligned(16))) _Float16 halo[10 * 10 * HID]; // 25600B
  __shared__ __attribute__((aligned(16))) _Float16 hhH[Nt * HID];       // 16384B
  __shared__ __attribute__((aligned(16))) _Float16 W2T[Cc * HID];       // [out][in]
  __shared__ float dwkL[9 * HID];
  __shared__ float dwbL[HID], b2L[Cc];

  const int tid = threadIdx.x, lane = tid & 31, wave = tid >> 5;
  const int row0 = wave << 4;
  const int bi = blockIdx.x;
  const int b  = bi >> 12;
  const int tidx = bi & 4095;
  const int y0 = (tidx >> 6) * WIN;
  const int x0 = (tidx & 63) * WIN;

  __builtin_prefetch(hIn + ((long long)b * (Hh * Ww) + (long long)y0 * Ww + x0) * HID, 0, 1);

  for (int i = tid; i < 9 * HID; i += 128) dwkL[i] = dwk[i];
  for (int i = tid; i < Cc * HID; i += 128)
    W2T[i] = (_Float16)W2[(i & 127) * Cc + (i >> 7)];
  dwbL[tid] = dwb[tid];
  if (tid < Cc) b2L[tid] = b2[tid];

  // ---- halo load: 1600 chunks of 16B, zero 'SAME' padding at image edges ---
  for (int i = tid; i < 100 * HID / 8; i += 128) {
    const int pix = i >> 4, part = i & 15;
    const int py = y0 + (pix / 10) - 1;
    const int px = x0 + (pix % 10) - 1;
    if (py >= 0 && py < Hh && px >= 0 && px < Ww) {
      const _Float16* gp =
          hIn + (((long long)b * (Hh * Ww) + (long long)py * Ww + px) * HID + part * 8);
#if ASYNC_COPY
      __builtin_amdgcn_global_load_async_to_lds_b128(
          (gvint4*)(uintptr_t)gp,
          (lvint4*)(unsigned)(uintptr_t)&halo[i * 8], 0, 0);
#else
      *(uint4*)&halo[i * 8] = *(const uint4*)gp;
#endif
    } else {
      const uint4 z = {0u, 0u, 0u, 0u};
      *(uint4*)&halo[i * 8] = z;
    }
  }
#if ASYNC_COPY
#if __has_builtin(__builtin_amdgcn_s_wait_asynccnt)
  __builtin_amdgcn_s_wait_asynccnt(0);
#else
  asm volatile("s_wait_asynccnt 0" ::: "memory");
#endif
#endif
  __syncthreads();

  // ---- depthwise 3x3 + bias + GELU, 8 channels per thread-iteration ----
  for (int i = tid; i < Nt * (HID / 8); i += 128) {   // 1024 chunks
    const int pix = i >> 4, cp = i & 15;
    const int py = pix >> 3, px = pix & 7;
    const int ch0 = cp * 8;
    float s[8];
#pragma unroll
    for (int e = 0; e < 8; ++e) s[e] = dwbL[ch0 + e];
#pragma unroll
    for (int ky = 0; ky < 3; ++ky)
#pragma unroll
      for (int kx = 0; kx < 3; ++kx) {
        const v8h hv = *(const v8h*)&halo[((py + ky) * 10 + (px + kx)) * HID + ch0];
#pragma unroll
        for (int e = 0; e < 8; ++e)
          s[e] += dwkL[(ky * 3 + kx) * HID + ch0 + e] * (float)hv[e];
      }
    v8h o;
#pragma unroll
    for (int e = 0; e < 8; ++e) o[e] = (_Float16)gelu_exact(s[e]);
    *(v8h*)&hhH[pix * HID + ch0] = o;
  }
  __syncthreads();

  // ---- W2 GEMM (K=128) + bias -> f32 staging (reuse halo region) ----
  float* outF = (float*)halo;   // halo dead after dwconv; 8192B needed
  const int fn = lane & 15, fh = lane >> 4;
#pragma unroll
  for (int nt = 0; nt < 2; ++nt) {
    v8f acc = {};
#pragma unroll
    for (int kc = 0; kc < 4; ++kc) {
      const v16h a  = ldA(&hhH[row0 * HID + kc * 32], HID, lane);
      const v16h bw = ldBT(&W2T[(nt * 16) * HID + kc * 32], HID, lane);
      acc = wmma16(a, bw, acc);
    }
    const int cidx = nt * 16 + fn;
#pragma unroll
    for (int r = 0; r < 8; ++r) {
      const int m = row0 + r + 8 * fh;
      outF[m * Cc + cidx] = acc[r] + b2L[cidx];
    }
  }
  __syncthreads();

  // ---- residual add + coalesced 16B stores ----
  for (int i = tid; i < Nt * Cc / 4; i += 128) {   // 512 float4
    const int t  = i >> 3, c4 = i & 7;
    const int py = y0 + (t >> 3), px = x0 + (t & 7);
    const long long gaddr =
        ((long long)b * (Hh * Ww) + (long long)py * Ww + px) * Cc + c4 * 4;
    float4 r4 = *(const float4*)&out[gaddr];          // x2 residual
    const float4 o4 = *(const float4*)&outF[t * Cc + c4 * 4];
    r4.x += o4.x; r4.y += o4.y; r4.z += o4.z; r4.w += o4.w;
    *(float4*)&out[gaddr] = r4;
  }
}

// ---------------------------------------------------------------------------
extern "C" void kernel_launch(void* const* d_in, const int* in_sizes, int n_in,
                              void* d_out, int out_size, void* d_ws,
                              size_t ws_size, hipStream_t stream) {
  const float* x          = (const float*)d_in[0];
  const float* g1         = (const float*)d_in[1];
  const float* beta1      = (const float*)d_in[2];
  const float* Wq         = (const float*)d_in[3];
  const float* bq         = (const float*)d_in[4];
  const float* Wkv        = (const float*)d_in[5];
  const float* bkv        = (const float*)d_in[6];
  const float* bias_table = (const float*)d_in[7];
  const float* Wp         = (const float*)d_in[8];
  const float* bp         = (const float*)d_in[9];
  const float* g2         = (const float*)d_in[10];
  const float* beta2      = (const float*)d_in[11];
  const float* W1         = (const float*)d_in[12];
  const float* b1m        = (const float*)d_in[13];
  const float* dw_k       = (const float*)d_in[14];
  const float* dw_b       = (const float*)d_in[15];
  const float* W2         = (const float*)d_in[16];
  const float* b2m        = (const float*)d_in[17];
  const int*   rel_idx    = (const int*)d_in[18];

  float*     out  = (float*)d_out;
  _Float16*  hbuf = (_Float16*)d_ws;   // B*H*W*HID f16 = 128 MiB

  const int nWin   = Bn * (Hh / WIN) * (Ww / WIN);   // 8192
  const int nTokBl = (Bn * Hh * Ww) / Nt;            // 8192

  kwin_attn<<<nWin, 128, 0, stream>>>(x, g1, beta1, Wq, bq, Wkv, bkv,
                                      bias_table, Wp, bp, rel_idx, out);
  kmlp1<<<nTokBl, 128, 0, stream>>>(out, g2, beta2, W1, b1m, hbuf);
  kmlp2<<<nWin, 128, 0, stream>>>(hbuf, dw_k, dw_b, W2, b2m, out);
}
